// LSALayer_71511205478733
// MI455X (gfx1250) — matrix operand order repaired and checked by reading, not compiled, not run on hardware
//
#include <hip/hip_runtime.h>
#include <stdint.h>

// ---------------------------------------------------------------------------
// MI455X (gfx1250) linear self-attention:  out = x + ((X Wq)(X Wk)^T (X Wv)) P
// Reassociated as Q (K^T V): attention core is D x D per batch, never S x S.
// f16-in / f32-accumulate via v_wmma_f32_16x16x32_f16 (wave32).
// Global -> LDS uses CDNA5 GLOBAL_LOAD_ASYNC_TO_LDS_B128 (ASYNCcnt) with
// double-buffered LDS so DMA of slab t+1 overlaps the 16 WMMAs of slab t.
// ---------------------------------------------------------------------------

typedef _Float16 f16;
typedef _Float16 f16x4 __attribute__((ext_vector_type(4)));
typedef _Float16 f16x8 __attribute__((ext_vector_type(8)));
typedef _Float16 v16h  __attribute__((ext_vector_type(16)));
typedef float    v8f   __attribute__((ext_vector_type(8)));

// ---------------- elementwise f32 -> f16 convert ---------------------------
__global__ __launch_bounds__(256) void cvt_f32_f16_kernel(
    const float4* __restrict__ in, f16x4* __restrict__ out, int n4) {
  int i = blockIdx.x * 256 + threadIdx.x;
  if (i < n4) {
    float4 v = in[i];
    f16x4 h = {(f16)v.x, (f16)v.y, (f16)v.z, (f16)v.w};
    out[i] = h;
  }
}

// ---------------- transpose + convert (D x D) ------------------------------
// out[n*D + k] = (f16) in[k*D + n]
__global__ __launch_bounds__(256) void transpose_cvt_kernel(
    const float* __restrict__ in, f16* __restrict__ out, int D) {
  __shared__ float tile[32][33];
  const int bx = blockIdx.x * 32;  // n-range
  const int by = blockIdx.y * 32;  // k-range
  const int tx = threadIdx.x;      // 0..31
  const int ty = threadIdx.y;      // 0..7
#pragma unroll
  for (int j = 0; j < 4; ++j)
    tile[ty + j * 8][tx] = in[(long)(by + ty + j * 8) * D + bx + tx];
  __syncthreads();
#pragma unroll
  for (int j = 0; j < 4; ++j)
    out[(long)(bx + ty + j * 8) * D + by + tx] = (f16)tile[tx][ty + j * 8];
}

// ---------------- NT GEMM: C[m][n] = sum_k A[m][k] * B[n][k] ---------------
// A: [M][K] (lda==K), B: [N][K] (ldb==K), C: [M][N] (ldc==N).
// M,N % 128 == 0, K % 64 == 0 (true for all calls: K in {1024, 4096}).
// 256 threads = 8 waves (4 M x 2 N); wave tile 32x64 = 2x4 WMMA accumulators.
constexpr int BM = 128, BN = 128, BK = 64;
constexpr int PITCH = 72;  // f16/row in LDS (144B: 16B-aligned, bank-spread)

// One CDNA5 async DMA: 32 lanes x 16B, global -> LDS, tracked by ASYNCcnt.
__device__ __forceinline__ void async_b128(unsigned lds_addr, const f16* g) {
  asm volatile("global_load_async_to_lds_b128 %0, %1, off"
               :: "v"(lds_addr), "v"(g)
               : "memory");
}

// Issue one 128x64 f16 slab (4 async b128 per thread-slot across the block).
__device__ __forceinline__ void issue_slab(const f16* __restrict__ gbase,
                                           int row0, int K, int k0,
                                           unsigned lds_base, int tid) {
#pragma unroll
  for (int i = 0; i < 4; ++i) {
    int id  = tid + i * 256;  // 0..1023 chunk id
    int row = id >> 3;        // 0..127
    int col = (id & 7) * 8;   // 0,8,...,56
    async_b128(lds_base + (unsigned)(row * PITCH + col) * 2u,
               gbase + (long)(row0 + row) * K + k0 + col);
  }
}

template <bool RESIDUAL>
__global__ __launch_bounds__(256) void gemm_nt_kernel(
    const f16* __restrict__ A, const f16* __restrict__ B,
    f16* __restrict__ C, float* __restrict__ Out, const float* __restrict__ Res,
    int M, int N, int K, long sA, long sB, long sC) {
  __shared__ alignas(16) f16 As[2][BM * PITCH];
  __shared__ alignas(16) f16 Bs[2][BN * PITCH];

  const int b = blockIdx.z;
  A += (long)b * sA;
  B += (long)b * sB;

  const int tid  = threadIdx.x;
  const int lane = tid & 31;
  const int wave = tid >> 5;
  const int wm   = wave & 3;   // wave row (0..3): 32 rows each
  const int wn   = wave >> 2;  // wave col (0..1): 64 cols each
  const int m0   = blockIdx.x * BM;
  const int n0   = blockIdx.y * BN;
  const int idx  = lane & 15;  // M (A frags) / N (B frags)
  const int hi   = lane >> 4;  // lane half selects K sub-range

  // LDS byte offsets (generic LDS pointer low 32 bits == LDS offset).
  const unsigned asBase = (unsigned)(uintptr_t)&As[0][0];
  const unsigned bsBase = (unsigned)(uintptr_t)&Bs[0][0];
  const unsigned bufStride = (unsigned)(BM * PITCH) * 2u;

  v8f acc[2][4];
#pragma unroll
  for (int mt = 0; mt < 2; ++mt)
#pragma unroll
    for (int nt = 0; nt < 4; ++nt) {
      v8f z = {0.f, 0.f, 0.f, 0.f, 0.f, 0.f, 0.f, 0.f};
      acc[mt][nt] = z;
    }

  const int nk = K / BK;
  // Prologue: slab 0 -> buffer 0 (8 async b128 per thread).
  issue_slab(A, m0, K, 0, asBase, tid);
  issue_slab(B, n0, K, 0, bsBase, tid);

  for (int t = 0; t < nk; ++t) {
    const int cur = t & 1;
    if (t + 1 < nk) {
      // DMA slab t+1 into the idle buffer; it overlaps this slab's WMMAs.
      issue_slab(A, m0, K, (t + 1) * BK, asBase + (cur ^ 1) * bufStride, tid);
      issue_slab(B, n0, K, (t + 1) * BK, bsBase + (cur ^ 1) * bufStride, tid);
      // In-order completion: <=8 outstanding  ==>  slab t (oldest 8) landed.
      asm volatile("s_wait_asynccnt 0x8" ::: "memory");
    } else {
      asm volatile("s_wait_asynccnt 0x0" ::: "memory");
    }
    __syncthreads();  // all waves' slab-t DMA visible

    const f16* a_lds = &As[cur][0];
    const f16* b_lds = &Bs[cur][0];
#pragma unroll
    for (int ks = 0; ks < 2; ++ks) {  // two K=32 sub-steps of the 64-slab
      // A frag (ISA 7.12.2, 16-bit A 16x32): v0-3 K=hi*8.., v4-7 K=16+hi*8..
      v16h af[2];
#pragma unroll
      for (int mt = 0; mt < 2; ++mt) {
        const f16* p = a_lds + (wm * 32 + mt * 16 + idx) * PITCH + ks * 32;
        f16x8 c0 = *(const f16x8*)(p + hi * 8);
        f16x8 c1 = *(const f16x8*)(p + 16 + hi * 8);
        af[mt] = __builtin_shufflevector(c0, c1, 0, 1, 2, 3, 4, 5, 6, 7, 8, 9,
                                         10, 11, 12, 13, 14, 15);
      }
      // B frag (same K order as A): v0-3 K=hi*8.., v4-7 K=16+hi*8..
      v16h bf[4];
#pragma unroll
      for (int nt = 0; nt < 4; ++nt) {
        const f16* p =
            b_lds + (wn * 64 + nt * 16 + idx) * PITCH + ks * 32;
        f16x8 c0 = *(const f16x8*)(p + hi * 8);
        f16x8 c1 = *(const f16x8*)(p + 16 + hi * 8);
        bf[nt] = __builtin_shufflevector(c0, c1, 0, 1, 2, 3, 4, 5, 6, 7, 8, 9,
                                         10, 11, 12, 13, 14, 15);
      }
#pragma unroll
      for (int mt = 0; mt < 2; ++mt)
#pragma unroll
        for (int nt = 0; nt < 4; ++nt)
          acc[mt][nt] = __builtin_amdgcn_wmma_f32_16x16x32_f16(
              false, af[mt], false, bf[nt], (short)0, acc[mt][nt], false,
              false);
    }
    __syncthreads();  // seal reads of buffer `cur` before t+1 overwrites it
  }

  // Epilogue. C layout: VGPR i, lane l -> row = 8*(l>>4)+i, col = l&15.
#pragma unroll
  for (int mt = 0; mt < 2; ++mt) {
    int r0 = m0 + wm * 32 + mt * 16 + hi * 8;
#pragma unroll
    for (int nt = 0; nt < 4; ++nt) {
      int c = n0 + wn * 64 + nt * 16 + idx;
#pragma unroll
      for (int i = 0; i < 8; ++i) {
        long off = (long)b * sC + (long)(r0 + i) * N + c;
        if constexpr (RESIDUAL)
          Out[off] = Res[off] + acc[mt][nt][i];
        else
          C[off] = (f16)acc[mt][nt][i];
      }
    }
  }
}

// ---------------------------------------------------------------------------
extern "C" void kernel_launch(void* const* d_in, const int* in_sizes, int n_in,
                              void* d_out, int out_size, void* d_ws,
                              size_t ws_size, hipStream_t stream) {
  (void)in_sizes; (void)n_in; (void)out_size; (void)ws_size;
  const int Bb = 4, S = 4096, D = 1024;
  const float* x  = (const float*)d_in[0];
  const float* Wq = (const float*)d_in[1];
  const float* Wk = (const float*)d_in[2];
  const float* Wv = (const float*)d_in[3];
  const float* P  = (const float*)d_in[4];
  float* out = (float*)d_out;

  const long SD = (long)S * D;  // 4M elems
  const long DD = (long)D * D;  // 1M elems

  f16* ws  = (f16*)d_ws;
  f16* x16 = ws;  ws += (long)Bb * SD;
  f16* WqT = ws;  ws += DD;
  f16* WkT = ws;  ws += DD;
  f16* WvT = ws;  ws += DD;
  f16* PT  = ws;  ws += DD;
  f16* Q16 = ws;  ws += (long)Bb * SD;
  f16* KT  = ws;  ws += (long)Bb * SD;  // reused as context after M^T GEMM
  f16* VT  = ws;  ws += (long)Bb * SD;
  f16* MT  = ws;  ws += (long)Bb * DD;
  f16* CT  = KT;                        // context aliases K^T (dead by then)

  // 1) x -> f16
  int n4 = (Bb * S * D) / 4;
  cvt_f32_f16_kernel<<<n4 / 256, 256, 0, stream>>>((const float4*)x,
                                                   (f16x4*)x16, n4);
  // 2) transposed f16 weights: WT[n][k] = W[k][n]
  dim3 tb(32, 8), tg(D / 32, D / 32);
  transpose_cvt_kernel<<<tg, tb, 0, stream>>>(Wq, WqT, D);
  transpose_cvt_kernel<<<tg, tb, 0, stream>>>(Wk, WkT, D);
  transpose_cvt_kernel<<<tg, tb, 0, stream>>>(Wv, WvT, D);
  transpose_cvt_kernel<<<tg, tb, 0, stream>>>(P, PT, D);

  dim3 blk(256);
  // 3) Q = X Wq                (A = x16 [s][d], B = WqT [n][d])
  gemm_nt_kernel<false><<<dim3(S / 128, D / 128, Bb), blk, 0, stream>>>(
      x16, WqT, Q16, nullptr, nullptr, S, D, D, SD, 0, SD);
  // 4) K^T = Wk^T X^T          (A = WkT [dout][din], B = x16 [s][din])
  gemm_nt_kernel<false><<<dim3(D / 128, S / 128, Bb), blk, 0, stream>>>(
      WkT, x16, KT, nullptr, nullptr, D, S, D, 0, SD, SD);
  // 5) V^T = Wv^T X^T
  gemm_nt_kernel<false><<<dim3(D / 128, S / 128, Bb), blk, 0, stream>>>(
      WvT, x16, VT, nullptr, nullptr, D, S, D, 0, SD, SD);
  // 6) M^T = V^T K             (A = VT [d2][s], B = KT [d1][s]), K-dim = S
  gemm_nt_kernel<false><<<dim3(D / 128, D / 128, Bb), blk, 0, stream>>>(
      VT, KT, MT, nullptr, nullptr, D, D, S, SD, SD, DD);
  // 7) context = Q M           (A = Q16 [s][d1], B = MT [d2][d1])
  gemm_nt_kernel<false><<<dim3(S / 128, D / 128, Bb), blk, 0, stream>>>(
      Q16, MT, CT, nullptr, nullptr, S, D, D, SD, DD, SD);
  // 8) out = x + context P     (A = CT [s][v], B = PT [o][v], f32 residual)
  gemm_nt_kernel<true><<<dim3(S / 128, D / 128, Bb), blk, 0, stream>>>(
      CT, PT, nullptr, out, x, S, D, D, SD, 0, SD);
}
